// AdaptiveFusion_88158498718228
// MI455X (gfx1250) — matrix operand order repaired and checked by reading, not compiled, not run on hardware
//
#include <hip/hip_runtime.h>
#include <hip/hip_bf16.h>
#include <math.h>

typedef _Float16 v16h __attribute__((ext_vector_type(16)));
typedef _Float16 v8h  __attribute__((ext_vector_type(8)));
typedef float    v8f  __attribute__((ext_vector_type(8)));

#define BB 4
#define CC 64
#define NPOS 4096          // 64*64
#define NMASK 512          // 8*64

// Inverse of the A-fragment K/ci permutation: value for index w goes to slot
// fraginv(w) so that a straight v16h load yields the hardware A layout
// ({base+0..7, base+16..23} per lane half).
__device__ __forceinline__ int fraginv(int w) {
    int lo = w & 7, t = w >> 3;
    int q  = ((t >> 2) << 1) | (t & 1);
    int h8 = (t >> 1) & 1;
    return q * 16 + h8 * 8 + lo;
}

// ---------------------------------------------------------------------------
// MaxPool2d(8,8) on mask + binarize -> mf[b*NPOS+p] in {0,1}
// ---------------------------------------------------------------------------
__global__ __launch_bounds__(256) void k_maskpool(const float* __restrict__ mask,
                                                  float* __restrict__ mf) {
    int idx = blockIdx.x * 256 + threadIdx.x;            // b*4096 + p
    int b = idx >> 12, p = idx & 4095;
    int oh = p >> 6, ow = p & 63;
    const float* mb = mask + (size_t)b * NMASK * NMASK;
    float mx = -INFINITY;
#pragma unroll
    for (int i = 0; i < 8; ++i) {
        const float* row = mb + (size_t)(oh * 8 + i) * NMASK + ow * 8;
#pragma unroll
        for (int j = 0; j < 8; ++j) mx = fmaxf(mx, row[j]);
    }
    mf[idx] = (mx > 0.f) ? 1.f : 0.f;
}

// ---------------------------------------------------------------------------
// Weights (OIHW 64x64x3x3) -> f16 Wt[pos][co][ci natural] for WMMA A frags.
// ---------------------------------------------------------------------------
__global__ __launch_bounds__(64) void k_wprep(const float* __restrict__ w,
                                              _Float16* __restrict__ Wt) {
    int idx = blockIdx.x * 64 + threadIdx.x;    // (pos, co), 576 total
    if (idx >= 576) return;
    int pos = idx >> 6, co = idx & 63;
    int kh = pos / 3, kw = pos % 3;
    _Float16* orow = Wt + (size_t)(pos * CC + co) * CC;
#pragma unroll
    for (int j = 0; j < CC; ++j) {
        int q = j >> 4, r = j & 15, h8 = r >> 3, lo = r & 7;
        int ci = j;   // natural ci
        orow[j] = (_Float16)w[((size_t)co * CC + ci) * 9 + kh * 3 + kw];
    }
}

// ---------------------------------------------------------------------------
// f32 plane-major (b,c,p) -> f16 position-major rows [b*4096+p][c] (natural ci)
// ---------------------------------------------------------------------------
__global__ __launch_bounds__(256) void k_to_t16(const float* __restrict__ src,
                                                _Float16* __restrict__ dst) {
    int idx = blockIdx.x * 256 + threadIdx.x;            // b*4096 + p
    const float* sp = src + ((size_t)(idx >> 12) * CC * NPOS) + (idx & 4095);
    _Float16* dp = dst + (size_t)idx * CC;
#pragma unroll 8
    for (int c = 0; c < CC; ++c) dp[c] = (_Float16)sp[(size_t)c * NPOS];
}

// ---------------------------------------------------------------------------
// 3x3 conv (pad=1) as 9 shifted 64x64xN WMMA GEMMs.
// Block = 4 waves; tile = 16 pixels of one row; wave w owns co [16w,16w+16).
// ---------------------------------------------------------------------------
__global__ __launch_bounds__(128) void k_conv3_wmma(const _Float16* __restrict__ xT,
                                                    const _Float16* __restrict__ Wt,
                                                    float* __restrict__ out) {
    const int b = blockIdx.x >> 8;
    const int tile = blockIdx.x & 255;
    const int pbase = tile << 4;
    const int h = pbase >> 6;                 // uniform row for whole tile
    const int wbase = pbase & 63;
    const int wid = threadIdx.x >> 5, lane = threadIdx.x & 31;
    const int n16 = lane & 15, hi = lane >> 4;
    const int goff = hi ? 16 : 0;
    const _Float16* xTb = xT + (size_t)b * NPOS * CC;

    v8f acc = {};
#pragma unroll
    for (int kh = 0; kh < 3; ++kh) {
        int hh = h + kh - 1;
        if ((unsigned)hh >= 64u) continue;    // uniform branch
#pragma unroll
        for (int kw = 0; kw < 3; ++kw) {
            int dw = kw - 1;
            int wc = wbase + n16 + dw;
            bool valid = ((unsigned)wc < 64u);
            int psrc = hh * 64 + (valid ? wc : (wbase + n16));
            const _Float16* xr = xTb + (size_t)psrc * CC;
            v16h b0 = *(const v16h*)(xr + goff);
            v16h b1 = *(const v16h*)(xr + 32 + goff);
            if (!valid) { v16h z = {}; b0 = z; b1 = z; }
            const _Float16* wr = Wt + (size_t)((kh * 3 + kw) * CC + 16 * wid + n16) * CC;
            v16h a0 = *(const v16h*)(wr + goff);
            v16h a1 = *(const v16h*)(wr + 32 + goff);
            acc = __builtin_amdgcn_wmma_f32_16x16x32_f16(false, a0, false, b0, (short)0, acc, false, false);
            acc = __builtin_amdgcn_wmma_f32_16x16x32_f16(false, a1, false, b1, (short)0, acc, false, false);
        }
    }
#pragma unroll
    for (int r = 0; r < 8; ++r) {
        int co = 16 * wid + r + (hi ? 8 : 0);
        out[((size_t)b * CC + co) * NPOS + pbase + n16] = acc[r];
    }
}

// ---------------------------------------------------------------------------
// InstanceNorm (biased var, eps=1e-5) [+ residual add] + ReLU.
// ---------------------------------------------------------------------------
__global__ __launch_bounds__(256) void k_inorm(const float* __restrict__ y,
                                               const float* __restrict__ xadd,
                                               float* __restrict__ out) {
    __shared__ float red[256], red2[256];
    int bc = blockIdx.x;
    const float* yp = y + (size_t)bc * NPOS;
    float s = 0.f, s2 = 0.f;
    for (int i = threadIdx.x; i < NPOS; i += 256) {
        float v = yp[i]; s += v; s2 = fmaf(v, v, s2);
    }
    red[threadIdx.x] = s; red2[threadIdx.x] = s2;
    __syncthreads();
    for (int st = 128; st > 0; st >>= 1) {
        if (threadIdx.x < st) {
            red[threadIdx.x] += red[threadIdx.x + st];
            red2[threadIdx.x] += red2[threadIdx.x + st];
        }
        __syncthreads();
    }
    float mu = red[0] * (1.f / NPOS);
    float var = red2[0] * (1.f / NPOS) - mu * mu;
    float inv = rsqrtf(var + 1e-5f);
    const float* xp = xadd ? xadd + (size_t)bc * NPOS : nullptr;
    float* op = out + (size_t)bc * NPOS;
    for (int i = threadIdx.x; i < NPOS; i += 256) {
        float v = (yp[i] - mu) * inv;
        if (xp) v += xp[i];
        op[i] = fmaxf(v, 0.f);
    }
}

// ---------------------------------------------------------------------------
// Build f16 Qt (natural rows), Kt (natural rows), Vh (c-major, natural k order).
// ---------------------------------------------------------------------------
__global__ __launch_bounds__(256) void k_prep(const float* __restrict__ kx,
                                              const float* __restrict__ x,
                                              const float* __restrict__ mf,
                                              _Float16* __restrict__ Qt,
                                              _Float16* __restrict__ Kt,
                                              _Float16* __restrict__ Vh) {
    int idx = blockIdx.x * 256 + threadIdx.x;            // b*4096 + p
    int b = idx >> 12, p = idx & 4095;
    const float* kb = kx + (size_t)b * CC * NPOS + p;
    const float* xb = x + (size_t)b * CC * NPOS + p;
    float s2 = 0.f;
#pragma unroll 4
    for (int c = 0; c < CC; ++c) { float v = kb[(size_t)c * NPOS]; s2 = fmaf(v, v, s2); }
    float nrm = sqrtf(s2);
    float mv = mf[idx], bk = 1.f - mv;
    float qs = mv / (mv * nrm + 1e-8f);
    float ks = bk / (bk * nrm + 1e-8f);
    _Float16* qrow = Qt + (size_t)idx * CC;
    _Float16* krow = Kt + (size_t)idx * CC;
    int pperm = p;                                       // natural k order
    _Float16* vcol = Vh + (size_t)b * CC * NPOS + pperm;
#pragma unroll 4
    for (int c = 0; c < CC; ++c) {
        float kv = kb[(size_t)c * NPOS];
        qrow[c] = (_Float16)(kv * qs);
        krow[c] = (_Float16)(kv * ks);
        vcol[(size_t)c * NPOS] = (_Float16)(xb[(size_t)c * NPOS] * bk);
    }
}

__global__ __launch_bounds__(256) void k_zero(float* __restrict__ p, int n) {
    int i = blockIdx.x * 256 + threadIdx.x;
    if (i < n) p[i] = 0.f;
}

// ---------------------------------------------------------------------------
// Masked cosine attention: single QK pass (P kept in 128KB LDS), PV pass,
// visatt column sums. Block = 4 waves = one 16-query tile.
// Scores bounded in [-20,20] -> exp(20s-20), no running max needed.
// ---------------------------------------------------------------------------
#define ATT_SMEM (16 * NPOS * 2 + 128)

struct alignas(128) SmemBlk { char b[128]; };

__global__ __launch_bounds__(128) void k_att(const _Float16* __restrict__ Qt,
                                             const _Float16* __restrict__ Kt,
                                             const _Float16* __restrict__ Vh,
                                             const float* __restrict__ mf,
                                             float* __restrict__ att_fore,
                                             float* __restrict__ visatt) {
    extern __shared__ SmemBlk smemblk[];
    _Float16* Pall = (_Float16*)smemblk;                       // [16][4096]
    float* Llds = (float*)((char*)smemblk + 16 * NPOS * 2);    // [16]
    float* scale_lds = Llds + 16;                              // [16]

    const int wid = threadIdx.x >> 5;
    const int lane = threadIdx.x & 31;
    const int n16 = lane & 15;
    const int hi = lane >> 4;
    const int off = hi ? 8 : 0;
    const int goff = hi ? 16 : 0;
    const int b = blockIdx.x >> 8;
    const int qbase = (blockIdx.x & 255) << 4;
    const float* mb = mf + b * NPOS;

    // ---- all-background q-tile: att == 0 exactly; write zeros and exit ----
    float qany = 0.f;
#pragma unroll
    for (int i = 0; i < 16; ++i) qany += mb[qbase + i];
    if (qany == 0.f) {
#pragma unroll
        for (int r = 0; r < 8; ++r) {
            int c = 16 * wid + r + off;
            att_fore[((size_t)b * CC + c) * NPOS + qbase + n16] = 0.f;
        }
        return;
    }

    const _Float16* Qtb = Qt + (size_t)b * NPOS * CC;
    const _Float16* Ktb = Kt + (size_t)b * NPOS * CC;
    const _Float16* Vhb = Vh + (size_t)b * CC * NPOS;

    // A fragments of Q^T: natural ci rows, contiguous k per lane half -> straight aligned loads.
    const _Float16* qrow = Qtb + (size_t)(qbase + n16) * CC;
    v16h aq0 = *(const v16h*)(qrow + goff);
    v16h aq1 = *(const v16h*)(qrow + 32 + goff);

    if (threadIdx.x < 16) Llds[threadIdx.x] = 0.f;
    __syncthreads();

    // ---- pass 1: QK WMMAs, exp, P -> LDS, accumulate L ----
    float Lp[8];
#pragma unroll
    for (int r = 0; r < 8; ++r) Lp[r] = 0.f;

    for (int it = 0; it < NPOS / 64; ++it) {
        int kg = it * 64 + wid * 16 + n16;
        const _Float16* krow = Ktb + (size_t)kg * CC;
        v16h bk0 = *(const v16h*)(krow + goff);
        v16h bk1 = *(const v16h*)(krow + 32 + goff);
        v8f s = {};
        s = __builtin_amdgcn_wmma_f32_16x16x32_f16(false, aq0, false, bk0, (short)0, s, false, false);
        s = __builtin_amdgcn_wmma_f32_16x16x32_f16(false, aq1, false, bk1, (short)0, s, false, false);
        float km = 1.f - mb[kg];
#pragma unroll
        for (int r = 0; r < 8; ++r) {
            float p = km * __expf(fmaf(20.f, s[r], -20.f));
            Lp[r] += p;
            Pall[(r + off) * NPOS + kg] = (_Float16)p;
        }
    }
#pragma unroll
    for (int r = 0; r < 8; ++r) {
        float v = Lp[r];
        v += __shfl_xor(v, 1, 16);
        v += __shfl_xor(v, 2, 16);
        v += __shfl_xor(v, 4, 16);
        v += __shfl_xor(v, 8, 16);
        if (n16 == 0) atomicAdd(&Llds[r + off], v);
    }
    __syncthreads();
    if (threadIdx.x < 16) {
        float L = Llds[threadIdx.x];
        float qm = mb[qbase + threadIdx.x];
        scale_lds[threadIdx.x] = (L > 0.f) ? (qm / L) : 0.f;
    }
    __syncthreads();

    // ---- pass 2: PV WMMAs straight from LDS; wave owns channels [16w,16w+16) ----
    v8f Oacc = {};
    for (int it = 0; it < NPOS / 64; ++it) {
        int kbase = it * 64;
        const _Float16* vrow = Vhb + (size_t)(16 * wid + n16) * NPOS + kbase;
        v16h av0 = *(const v16h*)(vrow + goff);          // natural k window, contiguous k per lane half
        v16h av1 = *(const v16h*)(vrow + 32 + goff);
        const _Float16* prow = Pall + n16 * NPOS + kbase;
        v16h bp0 = *(const v16h*)(prow + goff);
        v16h bp1 = *(const v16h*)(prow + 32 + goff);
        Oacc = __builtin_amdgcn_wmma_f32_16x16x32_f16(false, av0, false, bp0, (short)0, Oacc, false, false);
        Oacc = __builtin_amdgcn_wmma_f32_16x16x32_f16(false, av1, false, bp1, (short)0, Oacc, false, false);
    }
    float nscale = scale_lds[n16];
#pragma unroll
    for (int r = 0; r < 8; ++r) {
        int c = 16 * wid + r + off;
        att_fore[((size_t)b * CC + c) * NPOS + qbase + n16] = Oacc[r] * nscale;
    }

    // ---- pass 3: visatt column sums from LDS P ----
    float sc[16];
#pragma unroll
    for (int i = 0; i < 16; ++i) sc[i] = scale_lds[i];
    for (int k = threadIdx.x; k < NPOS; k += 128) {
        float accv = 0.f;
#pragma unroll
        for (int q = 0; q < 16; ++q)
            accv = fmaf((float)Pall[q * NPOS + k], sc[q], accv);
        atomicAdd(&visatt[b * NPOS + k], accv);
    }
}

// ---------------------------------------------------------------------------
// Fusion 1x1 convs
// ---------------------------------------------------------------------------
__global__ __launch_bounds__(256) void k_feat1(const float* __restrict__ x,
                                               const float* __restrict__ af,
                                               const float* __restrict__ mf,
                                               const float* __restrict__ w1,
                                               float* __restrict__ feat1) {
    __shared__ float wl[128];
    int bi = blockIdx.x;
    int chunk = bi & 15, co = (bi >> 4) & 63, b = bi >> 10;
    if (threadIdx.x < 128) wl[threadIdx.x] = w1[(size_t)co * 128 + threadIdx.x];
    __syncthreads();
    int p = chunk * 256 + threadIdx.x;
    float mv = mf[b * NPOS + p];
    const float* xb = x + (size_t)b * CC * NPOS + p;
    const float* ab = af + (size_t)b * CC * NPOS + p;
    float acc = 0.f;
#pragma unroll 4
    for (int ci = 0; ci < CC; ++ci) acc = fmaf(wl[ci], xb[(size_t)ci * NPOS], acc);
#pragma unroll 4
    for (int ci = 0; ci < CC; ++ci) acc = fmaf(wl[64 + ci], ab[(size_t)ci * NPOS], acc);
    feat1[((size_t)b * CC + co) * NPOS + p] = acc * mv;
}

__global__ __launch_bounds__(256) void k_gate(const float* __restrict__ feat1,
                                              const float* __restrict__ w2,
                                              const float* __restrict__ b2,
                                              float* __restrict__ sg) {
    __shared__ float wl[64];
    int bi = blockIdx.x;
    int chunk = bi & 15, co = (bi >> 4) & 63, b = bi >> 10;
    if (threadIdx.x < 64) wl[threadIdx.x] = w2[(size_t)co * 64 + threadIdx.x];
    __syncthreads();
    int p = chunk * 256 + threadIdx.x;
    const float* fb = feat1 + (size_t)b * CC * NPOS + p;
    float acc = b2[co];
#pragma unroll 4
    for (int ci = 0; ci < CC; ++ci) acc = fmaf(wl[ci], fb[(size_t)ci * NPOS], acc);
    sg[((size_t)b * CC + co) * NPOS + p] = 1.f / (1.f + __expf(-acc));
}

__global__ __launch_bounds__(256) void k_fused(const float* __restrict__ x,
                                               const float* __restrict__ af,
                                               const float* __restrict__ mf,
                                               const float* __restrict__ sg,
                                               const float* __restrict__ wfuse,
                                               float* __restrict__ out) {
    __shared__ float wl[128];
    int bi = blockIdx.x;
    int chunk = bi & 15, co = (bi >> 4) & 63, b = bi >> 10;
    if (threadIdx.x < 128) wl[threadIdx.x] = wfuse[(size_t)co * 128 + threadIdx.x];
    __syncthreads();
    int p = chunk * 256 + threadIdx.x;
    float mv = mf[b * NPOS + p];
    const float* xb = x + (size_t)b * CC * NPOS + p;
    const float* ab = af + (size_t)b * CC * NPOS + p;
    const float* sb = sg + (size_t)b * CC * NPOS + p;
    float acc = 0.f;
#pragma unroll 4
    for (int ci = 0; ci < CC; ++ci) acc = fmaf(wl[ci], xb[(size_t)ci * NPOS] * mv, acc);
#pragma unroll 4
    for (int ci = 0; ci < CC; ++ci)
        acc = fmaf(wl[64 + ci], ab[(size_t)ci * NPOS] * mv * (1.f - sb[(size_t)ci * NPOS]), acc);
    float xv = xb[(size_t)co * NPOS];
    out[((size_t)b * CC + co) * NPOS + p] = acc * mv + xv * (1.f - mv);
}

// ---------------------------------------------------------------------------
// Global max over visatt, then x8 nearest upsample + normalize.
// ---------------------------------------------------------------------------
__global__ __launch_bounds__(256) void k_vmax(const float* __restrict__ visatt,
                                              float* __restrict__ maxv) {
    __shared__ float red[256];
    float m = -INFINITY;
    for (int i = threadIdx.x; i < BB * NPOS; i += 256) m = fmaxf(m, visatt[i]);
    red[threadIdx.x] = m;
    __syncthreads();
    for (int st = 128; st > 0; st >>= 1) {
        if (threadIdx.x < st) red[threadIdx.x] = fmaxf(red[threadIdx.x], red[threadIdx.x + st]);
        __syncthreads();
    }
    if (threadIdx.x == 0) maxv[0] = red[0];
}

__global__ __launch_bounds__(256) void k_attmask(const float* __restrict__ visatt,
                                                 const float* __restrict__ maxv,
                                                 float* __restrict__ am) {
    int idx = blockIdx.x * 256 + threadIdx.x;   // b*512*512 + y*512 + x
    int b = idx >> 18;
    int rem = idx & 262143;
    int y = rem >> 9, x = rem & 511;
    float inv = 1.f / maxv[0];
    am[idx] = visatt[b * NPOS + (y >> 3) * 64 + (x >> 3)] * inv;
}

// ---------------------------------------------------------------------------
extern "C" void kernel_launch(void* const* d_in, const int* in_sizes, int n_in,
                              void* d_out, int out_size, void* d_ws, size_t ws_size,
                              hipStream_t stream) {
    const float* x     = (const float*)d_in[0];
    const float* mask  = (const float*)d_in[1];
    const float* kw1   = (const float*)d_in[2];
    const float* kw2   = (const float*)d_in[3];
    const float* fsw1  = (const float*)d_in[4];
    const float* fsw2  = (const float*)d_in[5];
    const float* fsb2  = (const float*)d_in[6];
    const float* fsfw  = (const float*)d_in[7];
    float* out = (float*)d_out;                       // out: 4*64*64*64 floats
    float* attmask = out + (size_t)BB * CC * NPOS;    // attmask: 4*512*512 floats

    char* ws = (char*)d_ws;
    size_t off = 0;
    auto carve = [&](size_t bytes) {
        char* p = ws + off;
        off += (bytes + 255) & ~(size_t)255;
        return p;
    };
    const size_t PLANE = (size_t)BB * CC * NPOS * sizeof(float);     // 4 MB
    const size_t HPLANE = (size_t)BB * CC * NPOS * sizeof(_Float16); // 2 MB
    float*    mf       = (float*)carve((size_t)BB * NPOS * sizeof(float));
    float*    y1       = (float*)carve(PLANE);
    float*    y1n      = (float*)carve(PLANE);
    float*    y2       = (float*)carve(PLANE);
    float*    kx       = (float*)carve(PLANE);
    _Float16* xT       = (_Float16*)carve(HPLANE);
    _Float16* yT       = (_Float16*)carve(HPLANE);
    _Float16* Wt1      = (_Float16*)carve((size_t)9 * CC * CC * sizeof(_Float16));
    _Float16* Wt2      = (_Float16*)carve((size_t)9 * CC * CC * sizeof(_Float16));
    _Float16* Qt       = (_Float16*)carve(HPLANE);
    _Float16* Kt       = (_Float16*)carve(HPLANE);
    _Float16* Vh       = (_Float16*)carve(HPLANE);
    float*    att_fore = (float*)carve(PLANE);
    float*    visatt   = (float*)carve((size_t)BB * NPOS * sizeof(float));
    float*    feat1    = (float*)carve(PLANE);
    float*    sg       = (float*)carve(PLANE);
    float*    maxv     = (float*)carve(256);
    (void)ws_size; (void)in_sizes; (void)n_in; (void)out_size;

    k_maskpool<<<BB * NPOS / 256, 256, 0, stream>>>(mask, mf);
    k_wprep<<<9, 64, 0, stream>>>(kw1, Wt1);
    k_wprep<<<9, 64, 0, stream>>>(kw2, Wt2);
    k_to_t16<<<BB * NPOS / 256, 256, 0, stream>>>(x, xT);
    k_conv3_wmma<<<BB * 256, 128, 0, stream>>>(xT, Wt1, y1);
    k_inorm<<<BB * CC, 256, 0, stream>>>(y1, nullptr, y1n);
    k_to_t16<<<BB * NPOS / 256, 256, 0, stream>>>(y1n, yT);
    k_conv3_wmma<<<BB * 256, 128, 0, stream>>>(yT, Wt2, y2);
    k_inorm<<<BB * CC, 256, 0, stream>>>(y2, x, kx);
    k_prep<<<BB * NPOS / 256, 256, 0, stream>>>(kx, x, mf, Qt, Kt, Vh);
    k_zero<<<BB * NPOS / 256, 256, 0, stream>>>(visatt, BB * NPOS);
    k_att<<<BB * (NPOS / 16), 128, ATT_SMEM, stream>>>(Qt, Kt, Vh, mf, att_fore, visatt);
    k_feat1<<<BB * CC * 16, 256, 0, stream>>>(x, att_fore, mf, fsw1, feat1);
    k_gate<<<BB * CC * 16, 256, 0, stream>>>(feat1, fsw2, fsb2, sg);
    k_fused<<<BB * CC * 16, 256, 0, stream>>>(x, att_fore, mf, sg, fsfw, out);
    k_vmax<<<1, 256, 0, stream>>>(visatt, maxv);
    k_attmask<<<BB * NMASK * NMASK / 256, 256, 0, stream>>>(visatt, maxv, attmask);
}
